// selfattention_7069516169559
// MI455X (gfx1250) — matrix operand order repaired and checked by reading, not compiled, not run on hardware
//
#include <hip/hip_runtime.h>

#define DEV __device__ __forceinline__

typedef __attribute__((ext_vector_type(16))) __bf16 v16bf;
typedef __attribute__((ext_vector_type(8)))  float  v8f;

constexpr int kB   = 8;
constexpr int kC   = 256;
constexpr int kCqk = 64;
constexpr int kN   = 64 * 64;   // 4096

// ---- CDNA5 async global->LDS path (probe-verified in round 3) --------------
#if defined(__has_builtin)
#if __has_builtin(__builtin_amdgcn_global_load_async_to_lds_b128) && \
    __has_builtin(__builtin_amdgcn_s_wait_asynccnt)
#define USE_ASYNC_LDS 1
#endif
#if __has_builtin(__builtin_amdgcn_tensor_load_to_lds) && \
    __has_builtin(__builtin_amdgcn_s_wait_tensorcnt)
#define HAVE_TDM 1
#endif
#endif
#ifndef USE_ASYNC_LDS
#define USE_ASYNC_LDS 0
#endif
#ifndef HAVE_TDM
#define HAVE_TDM 0
#endif

#if USE_ASYNC_LDS
// Probe-derived prototype: parameters are generic pointers to 'int __vector(4)'.
typedef __attribute__((__vector_size__(4 * sizeof(int)))) int v4i_ls;
DEV void async_cp16(const void* g, void* l) {
  __builtin_amdgcn_global_load_async_to_lds_b128((v4i_ls*)(void*)g, (v4i_ls*)l, 0, 0);
}
#define WAIT_ASYNC(n) __builtin_amdgcn_s_wait_asynccnt(n)
#else
#define WAIT_ASYNC(n)
#endif

#if HAVE_TDM
// gcc-style vectors to match builtin prototypes (cf. round-2 diagnostic style)
typedef unsigned tdm_v4u __attribute__((__vector_size__(16)));
typedef int      tdm_v8i __attribute__((__vector_size__(32)));
typedef int      tdm_v4i __attribute__((__vector_size__(16)));

// Build a 2D-tile D# (groups 0/1 per CDNA5 ISA) and issue
// TENSOR_LOAD_TO_LDS. Strides/dims in data_size (=2 byte) units.
DEV void tdm_load_2d(unsigned lds_off, const void* gptr,
                     unsigned tile_d0, unsigned tile_d1,
                     unsigned tensor_d0, unsigned tensor_d1,
                     unsigned long long stride_d0) {
  unsigned long long ga = (unsigned long long)(size_t)gptr;
  tdm_v4u g0;
  g0[0] = 1u;                                          // count=1, user descriptor
  g0[1] = lds_off;                                     // lds_addr (bytes)
  g0[2] = (unsigned)(ga & 0xffffffffu);                // global_addr[31:0]
  g0[3] = (unsigned)((ga >> 32) & 0x01ffffffu) | (2u << 30);  // addr[56:32] | type=2
  tdm_v8i g1;
  g1[0] = (int)(1u << 16);                             // wg_mask=0, data_size=2B
  g1[1] = (int)((tensor_d0 & 0xffffu) << 16);          // tensor_dim0[15:0]
  g1[2] = (int)((tensor_d0 >> 16) & 0xffffu)           // tensor_dim0[31:16]
        | (int)((tensor_d1 & 0xffffu) << 16);          // tensor_dim1[15:0]
  g1[3] = (int)((tensor_d1 >> 16) & 0xffffu)           // tensor_dim1[31:16]
        | (int)((tile_d0 & 0xffffu) << 16);            // tile_dim0
  g1[4] = (int)(tile_d1 & 0xffffu);                    // tile_dim1 (tile_dim2=0)
  g1[5] = (int)(stride_d0 & 0xffffffffu);              // dim0_stride[31:0]
  g1[6] = (int)((stride_d0 >> 32) & 0xffffu);          // dim0_stride[47:32]
  g1[7] = 0;                                           // dim1_stride=0 (2D)
  tdm_v4i z4 = {0, 0, 0, 0};                           // groups 2/3: tile_dim3/4=0
#if __clang_major__ >= 23
  tdm_v8i z8 = {0, 0, 0, 0, 0, 0, 0, 0};
  __builtin_amdgcn_tensor_load_to_lds(g0, g1, z4, z4, z8, 0);
#else
  __builtin_amdgcn_tensor_load_to_lds(g0, g1, z4, z4, 0);
#endif
}
#endif

// Newest-batch allowance for the V (and K, if not TDM) async loads per tile.
#if HAVE_TDM
#define ASYNC_KEEP 8
#else
#define ASYNC_KEEP 10
#endif

union FragU {
  unsigned u[8];
  v16bf    v;
};

DEV unsigned pack2_bf16(float a, float b) {
  union { __bf16 h[2]; unsigned u; } z;
  z.h[0] = (__bf16)a;
  z.h[1] = (__bf16)b;
  return z.u;
}

DEV v8f wmma_bf16(v16bf a, v16bf b, v8f c) {
  return __builtin_amdgcn_wmma_f32_16x16x32_bf16(false, a, false, b, (short)0, c,
                                                 false, false);
}

// ---------------------------------------------------------------------------
// QKV projection: out[o,n] = sum_c W[o,c] * x[b,c,n] + bias[o], stored bf16.
// ---------------------------------------------------------------------------
__global__ void __launch_bounds__(128)
qkv_proj_kernel(const float* __restrict__ Wm, const float* __restrict__ bias,
                const float* __restrict__ x, __bf16* __restrict__ outb,
                int Odim, int ch_major) {
  const int lane  = threadIdx.x & 31;
  const int wave  = threadIdx.x >> 5;
  const int hfl   = lane >> 4;
  const int lcol  = lane & 15;
  const int kbase = hfl * 8;
  const int b  = blockIdx.z;
  const int o0 = blockIdx.y * 16;
  const int n0 = blockIdx.x * 64 + wave * 16;

  const float* xb = x + (size_t)b * kC * kN;
  v8f acc = {};
  for (int c0 = 0; c0 < kC; c0 += 32) {
    FragU afr, bfr;
    const float* wrow = Wm + (size_t)(o0 + lcol) * kC + c0 + kbase;
#pragma unroll
    for (int r = 0; r < 4; ++r) {
      float2 w0 = *(const float2*)(wrow + 2 * r);
      float2 w1 = *(const float2*)(wrow + 16 + 2 * r);
      afr.u[r]     = pack2_bf16(w0.x, w0.y);
      afr.u[4 + r] = pack2_bf16(w1.x, w1.y);
    }
    const float* xcol = xb + (size_t)(c0 + kbase) * kN + n0 + lcol;
#pragma unroll
    for (int r = 0; r < 4; ++r) {
      bfr.u[r]     = pack2_bf16(xcol[(size_t)(2 * r) * kN], xcol[(size_t)(2 * r + 1) * kN]);
      bfr.u[4 + r] = pack2_bf16(xcol[(size_t)(16 + 2 * r) * kN], xcol[(size_t)(17 + 2 * r) * kN]);
    }
    acc = wmma_bf16(afr.v, bfr.v, acc);
  }
#pragma unroll
  for (int r = 0; r < 8; ++r) {
    int o = o0 + r + 8 * hfl;
    int n = n0 + lcol;
    float v = acc[r] + bias[o];
    if (ch_major)
      outb[((size_t)b * Odim + o) * kN + n] = (__bf16)v;
    else
      outb[((size_t)b * kN + n) * Odim + o] = (__bf16)v;
  }
}

// ---------------------------------------------------------------------------
// Flash attention. K tile staged by TDM (1 descriptor, wave 0), V tile by
// async global->LDS B128 (all lanes); both double-buffered.
// ---------------------------------------------------------------------------
constexpr int KS_OFF = 0;       // K tiles: 2 x [64 ch][32 j]  bf16 = 2 x 4 KB
constexpr int VS_OFF = 8192;    // V tiles: 2 x [32 j][256 c]  bf16 = 2 x 16 KB
constexpr int PS_OFF = 40960;   // P stage: 4 x [16][32] bf16 = 4 KB
constexpr int SMEM_BYTES = 45056;
// epilogue reuses smem[0..33792) as float [256][33]

DEV void stage_k(const __bf16* Kb, int j0, unsigned char* smem, int buf,
                 int tid, int wave) {
  unsigned char* ksd = smem + KS_OFF + buf * 4096;
#if HAVE_TDM
  if (wave == 0)
    tdm_load_2d((unsigned)(size_t)ksd, Kb + j0,
                /*tile_d0=*/32, /*tile_d1=*/64,
                /*tensor_d0=*/kN, /*tensor_d1=*/kCqk, /*stride_d0=*/kN);
#elif USE_ASYNC_LDS
#pragma unroll
  for (int it = 0; it < 2; ++it) {
    int idx = tid + it * 128;
    int row = idx >> 2, seg = idx & 3;
    async_cp16(Kb + (size_t)row * kN + j0 + seg * 8, ksd + idx * 16);
  }
#else
  uint4* ks4 = (uint4*)ksd;
  for (int idx = tid; idx < 256; idx += 128) {
    int row = idx >> 2, seg = idx & 3;
    ks4[idx] = *(const uint4*)(Kb + (size_t)row * kN + j0 + seg * 8);
  }
#endif
}

DEV void stage_v(const __bf16* Vb, int j0, unsigned char* smem, int buf, int tid) {
  unsigned char* vsd = smem + VS_OFF + buf * 16384;
#if USE_ASYNC_LDS
#pragma unroll
  for (int it = 0; it < 8; ++it) {
    int idx = tid + it * 128;
    int row = idx >> 5, seg = idx & 31;
    async_cp16(Vb + (size_t)(j0 + row) * kC + seg * 8, vsd + idx * 16);
  }
#else
  uint4* vs4 = (uint4*)vsd;
  for (int idx = tid; idx < 1024; idx += 128) {
    int row = idx >> 5, seg = idx & 31;
    vs4[idx] = *(const uint4*)(Vb + (size_t)(j0 + row) * kC + seg * 8);
  }
#endif
}

__global__ void __launch_bounds__(128)
flash_attn_kernel(const __bf16* __restrict__ Qt, const __bf16* __restrict__ Kc,
                  const __bf16* __restrict__ Vt, const float* __restrict__ xin,
                  const float* __restrict__ gamma, float* __restrict__ out) {
  __shared__ __align__(16) unsigned char smem[SMEM_BYTES];

  const int tid    = threadIdx.x;
  const int lane   = tid & 31;
  const int wave   = tid >> 5;
  const int hfl    = lane >> 4;
  const int lcol   = lane & 15;
  const int kbase2 = hfl * 4;

  const int b  = blockIdx.y;
  const int i0 = blockIdx.x * 64;
  const int iw = i0 + wave * 16;

  // Q A-fragments, register-resident for the whole j-loop.
  FragU qa0, qa1;
  {
    const unsigned* qu = (const unsigned*)(Qt + ((size_t)b * kN + iw + lcol) * kCqk);
#pragma unroll
    for (int r = 0; r < 4; ++r) {
      qa0.u[r]     = qu[kbase2 + r];
      qa0.u[4 + r] = qu[8 + kbase2 + r];
      qa1.u[r]     = qu[16 + kbase2 + r];
      qa1.u[4 + r] = qu[24 + kbase2 + r];
    }
  }

  v8f acc[16];
#pragma unroll
  for (int t = 0; t < 16; ++t) acc[t] = {};
  float m_run[8], l_run[8];
#pragma unroll
  for (int r = 0; r < 8; ++r) { m_run[r] = -INFINITY; l_run[r] = 0.f; }

  const __bf16* Kb = Kc + (size_t)b * kCqk * kN;
  const __bf16* Vb = Vt + (size_t)b * kN * kC;

  constexpr int kTiles = kN / 32;   // 128
  stage_k(Kb, 0, smem, 0, tid, wave);
  stage_v(Vb, 0, smem, 0, tid);

  for (int jt = 0; jt < kTiles; ++jt) {
    const int buf = jt & 1;
    if (jt + 1 < kTiles) {
      stage_k(Kb, (jt + 1) * 32, smem, buf ^ 1, tid, wave);
      stage_v(Vb, (jt + 1) * 32, smem, buf ^ 1, tid);
      WAIT_ASYNC(ASYNC_KEEP);     // newest batch (next tile) may stay in flight
#if HAVE_TDM
      if (wave == 0) __builtin_amdgcn_s_wait_tensorcnt(1);
#endif
    } else {
      WAIT_ASYNC(0);
#if HAVE_TDM
      if (wave == 0) __builtin_amdgcn_s_wait_tensorcnt(0);
#endif
    }
    __syncthreads();    // current tile visible to all waves

    const unsigned char* ks = smem + KS_OFF + buf * 4096;
    const unsigned char* vs = smem + VS_OFF + buf * 16384;

    // ---- S = Q^T K : two 16x16 j-tiles, K-dim = 64 channels
    v8f S0 = {}, S1 = {};
    {
      const __bf16* kh = (const __bf16*)ks;
      FragU kb;
#pragma unroll
      for (int r = 0; r < 16; ++r) kb.v[r] = kh[(hfl * 8 + (r & 7) + ((r >> 3) << 4)) * 32 + lcol];
      S0 = wmma_bf16(qa0.v, kb.v, S0);
#pragma unroll
      for (int r = 0; r < 16; ++r) kb.v[r] = kh[(hfl * 8 + (r & 7) + ((r >> 3) << 4)) * 32 + 16 + lcol];
      S1 = wmma_bf16(qa0.v, kb.v, S1);
#pragma unroll
      for (int r = 0; r < 16; ++r) kb.v[r] = kh[(32 + hfl * 8 + (r & 7) + ((r >> 3) << 4)) * 32 + lcol];
      S0 = wmma_bf16(qa1.v, kb.v, S0);
#pragma unroll
      for (int r = 0; r < 16; ++r) kb.v[r] = kh[(32 + hfl * 8 + (r & 7) + ((r >> 3) << 4)) * 32 + 16 + lcol];
      S1 = wmma_bf16(qa1.v, kb.v, S1);
    }

    // ---- online softmax
    float alpha[8];
#pragma unroll
    for (int r = 0; r < 8; ++r) {
      float mx = fmaxf(S0[r], S1[r]);
#pragma unroll
      for (int s = 1; s < 16; s <<= 1) mx = fmaxf(mx, __shfl_xor(mx, s, 32));
      float mn = fmaxf(m_run[r], mx);
      float a  = __expf(m_run[r] - mn);
      float e0 = __expf(S0[r] - mn);
      float e1 = __expf(S1[r] - mn);
      float rs = e0 + e1;
#pragma unroll
      for (int s = 1; s < 16; s <<= 1) rs += __shfl_xor(rs, s, 32);
      l_run[r] = l_run[r] * a + rs;
      m_run[r] = mn;
      alpha[r] = a;
      S0[r] = e0;
      S1[r] = e1;
    }
#pragma unroll
    for (int t = 0; t < 16; ++t)
#pragma unroll
      for (int r = 0; r < 8; ++r) acc[t][r] *= alpha[r];

    // ---- P: C/D layout -> A layout via per-wave LDS staging
    __bf16* Pw = (__bf16*)(smem + PS_OFF) + wave * 512;
#pragma unroll
    for (int r = 0; r < 8; ++r) {
      int row = r + 8 * hfl;
      Pw[row * 32 + lcol]      = (__bf16)S0[r];
      Pw[row * 32 + 16 + lcol] = (__bf16)S1[r];
    }
    __syncthreads();
    FragU pa;
    {
      const unsigned* pu = (const unsigned*)Pw + lcol * 16;
#pragma unroll
      for (int r = 0; r < 4; ++r) {
        pa.u[r]     = pu[kbase2 + r];
        pa.u[4 + r] = pu[8 + kbase2 + r];
      }
    }

    // ---- O += P x V : 16 c-tiles
    const __bf16* vh = (const __bf16*)vs;
#pragma unroll
    for (int t = 0; t < 16; ++t) {
      FragU vb;
#pragma unroll
      for (int r = 0; r < 16; ++r) vb.v[r] = vh[(hfl * 8 + (r & 7) + ((r >> 3) << 4)) * 256 + t * 16 + lcol];
      acc[t] = wmma_bf16(pa.v, vb.v, acc[t]);
    }
    __syncthreads();  // all waves done with `buf` before it is restaged
  }

  // ---- epilogue: out[b,c,i] = gamma * (O/l) + x, coalesced via LDS transpose
  float invl[8];
#pragma unroll
  for (int r = 0; r < 8; ++r) invl[r] = 1.f / l_run[r];
  const float gam = gamma[0];
  float* Osf = (float*)smem;   // [256][33] f32
  for (int p = 0; p < 2; ++p) {
    __syncthreads();
    if ((wave >> 1) == p) {
      int iloc = (wave & 1) * 16 + 8 * hfl;
#pragma unroll
      for (int t = 0; t < 16; ++t) {
        int c = t * 16 + lcol;
#pragma unroll
        for (int r = 0; r < 8; ++r) Osf[c * 33 + iloc + r] = acc[t][r] * invl[r];
      }
    }
    __syncthreads();
    for (int e = tid; e < 256 * 32; e += 128) {
      int c = e >> 5, ii = e & 31;
      size_t g = ((size_t)b * kC + c) * kN + i0 + p * 32 + ii;
      out[g] = gam * Osf[c * 33 + ii] + xin[g];
    }
  }
}

// ---------------------------------------------------------------------------
extern "C" void kernel_launch(void* const* d_in, const int* in_sizes, int n_in,
                              void* d_out, int out_size, void* d_ws, size_t ws_size,
                              hipStream_t stream) {
  const float* x     = (const float*)d_in[0];
  const float* Wq    = (const float*)d_in[1];
  const float* bq    = (const float*)d_in[2];
  const float* Wk    = (const float*)d_in[3];
  const float* bk    = (const float*)d_in[4];
  const float* Wv    = (const float*)d_in[5];
  const float* bv    = (const float*)d_in[6];
  const float* gamma = (const float*)d_in[7];
  float* out = (float*)d_out;

  // Workspace (bf16): Q^T [B][N][64] | K [B][64][N] | V^T [B][N][256]
  __bf16* Qt = (__bf16*)d_ws;
  __bf16* Kc = Qt + (size_t)kB * kN * kCqk;
  __bf16* Vt = Kc + (size_t)kB * kN * kCqk;

  dim3 blk(128);
  qkv_proj_kernel<<<dim3(kN / 64, kCqk / 16, kB), blk, 0, stream>>>(Wq, bq, x, Qt, kCqk, 0);
  qkv_proj_kernel<<<dim3(kN / 64, kCqk / 16, kB), blk, 0, stream>>>(Wk, bk, x, Kc, kCqk, 1);
  qkv_proj_kernel<<<dim3(kN / 64, kC / 16,   kB), blk, 0, stream>>>(Wv, bv, x, Vt, kC, 0);
  flash_attn_kernel<<<dim3(kN / 64, kB), blk, 0, stream>>>(Qt, Kc, Vt, x, gamma, out);
}
